// CIFLayer_48704929137286
// MI455X (gfx1250) — compile-verified, matrix operand order checked by reading, not run on hardware
//
#include <hip/hip_runtime.h>
#include <hip/hip_bf16.h>

// ---------------------------------------------------------------------------
// CIF layer forward for MI455X (gfx1250, wave32).
// S=4096, B=16, C=512, T_MAX=512, BETA=1.0
// Outputs: out (T, B, C) f32  then  alpha_sum (B,) f32, concatenated in d_out.
// ---------------------------------------------------------------------------

#define S_DIM 4096
#define B_DIM 16
#define C_DIM 512
#define T_MAX 512

typedef __attribute__((ext_vector_type(16))) __bf16 v16bf;
typedef __attribute__((ext_vector_type(8)))  float  v8f;
typedef __attribute__((ext_vector_type(4)))  int    v4i;

union BF16x16 {
    v16bf v;
    unsigned short u[16];
};

// Split fp32 into bf16 hi + bf16 lo (truncation split; lo captures residual).
static __device__ inline void split_bf16(float x, unsigned short& hi, unsigned short& lo) {
    union { float f; unsigned u; } a; a.f = x;
    hi = (unsigned short)(a.u >> 16);
    union { float f; unsigned u; } h; h.u = ((unsigned)hi) << 16;
    union { float f; unsigned u; } r; r.f = x - h.f;
    lo = (unsigned short)(r.u >> 16);
}

// ---------------------------------------------------------------------------
// CDNA5 async global->LDS staging (ASYNCcnt path), with safe fallback.
// Builtin signature (probe-confirmed via diagnostic): 
//   void __builtin_amdgcn_global_load_async_to_lds_b128(
//       v4i addrspace(1)* src, v4i addrspace(3)* dst, imm int off, imm int cpol)
// ---------------------------------------------------------------------------
#if defined(__has_builtin)
#if __has_builtin(__builtin_amdgcn_global_load_async_to_lds_b128)
#define USE_ASYNC_LDS 1
#endif
#endif
#ifndef USE_ASYNC_LDS
#define USE_ASYNC_LDS 0
#endif

static __device__ inline void copy16_async(const void* g, void* l) {
#if USE_ASYNC_LDS
    __builtin_amdgcn_global_load_async_to_lds_b128(
        (__attribute__((address_space(1))) v4i*)g,
        (__attribute__((address_space(3))) v4i*)l, 0, 0);
#else
    *(uint4*)l = *(const uint4*)g;
#endif
}

static __device__ inline void async_copy_wait() {
#if USE_ASYNC_LDS
    asm volatile("s_wait_asynccnt 0x0" ::: "memory");
#endif
}

// Load one 16-bf16 WMMA fragment from an LDS row laid out [row][k] contiguous:
// lane's 16 values are K = {half..half+7, 16+half..16+half+7} -> two b128 loads.
static __device__ inline void load_frag(BF16x16& f, const unsigned short* base, int half) {
    *(uint4*)&f.u[0] = *(const uint4*)(base + half);
    *(uint4*)&f.u[8] = *(const uint4*)(base + 16 + half);
}

// ---------------------------------------------------------------------------
// Kernel 1: alpha = sigmoid(masked last channel); alpha_sum per batch.
// ---------------------------------------------------------------------------
__global__ __launch_bounds__(256) void cif_alpha(
        const float* __restrict__ x, const unsigned char* __restrict__ mask,
        float* __restrict__ alpha_raw, float* __restrict__ asum,
        float* __restrict__ out_asum) {
    __shared__ float red[256];
    const int b = blockIdx.x;
    const int tid = threadIdx.x;
    float sum = 0.f;
    for (int s = tid; s < S_DIM; s += 256) {
        float a = x[((size_t)s * B_DIM + b) * C_DIM + (C_DIM - 1)];
        if (mask[b * S_DIM + s]) a = -10000.0f;
        float al = 1.0f / (1.0f + expf(-a));
        alpha_raw[b * S_DIM + s] = al;
        sum += al;
    }
    red[tid] = sum;
    __syncthreads();
    for (int off = 128; off > 0; off >>= 1) {
        if (tid < off) red[tid] += red[tid + off];
        __syncthreads();
    }
    if (tid == 0) { asum[b] = red[0]; out_asum[b] = red[0]; }
}

// ---------------------------------------------------------------------------
// Kernel 2: normalize alpha by tl/alpha_sum and cumsum over S (per batch).
// ---------------------------------------------------------------------------
__global__ __launch_bounds__(1024) void cif_scan(
        const float* __restrict__ alpha_raw, const float* __restrict__ asum,
        const int* __restrict__ tl, float* __restrict__ csum) {
    __shared__ float warr[1024];
    const int b = blockIdx.x;
    const int tid = threadIdx.x;
    const float scale = (float)tl[b] / asum[b];
    const int s0 = tid * 4;
    float v[4];
    float run = 0.f;
    #pragma unroll
    for (int j = 0; j < 4; ++j) {
        run += alpha_raw[b * S_DIM + s0 + j] * scale;
        v[j] = run;
    }
    warr[tid] = run;
    const float total = run;
    __syncthreads();
    for (int off = 1; off < 1024; off <<= 1) {
        float t = (tid >= off) ? warr[tid - off] : 0.f;
        __syncthreads();
        warr[tid] += t;
        __syncthreads();
    }
    const float excl = warr[tid] - total;
    #pragma unroll
    for (int j = 0; j < 4; ++j)
        csum[b * S_DIM + s0 + j] = excl + v[j];
}

// ---------------------------------------------------------------------------
// Kernel 3: zero padded feats (B, T, 512).
// ---------------------------------------------------------------------------
__global__ __launch_bounds__(256) void cif_zero(float4* __restrict__ p, int n4) {
    int i = blockIdx.x * 256 + threadIdx.x;
    if (i < n4) p[i] = make_float4(0.f, 0.f, 0.f, 0.f);
}

// ---------------------------------------------------------------------------
// Kernel 4: add-scatter of attn weights * x into feats (one wave per (b,s)).
// ---------------------------------------------------------------------------
__global__ __launch_bounds__(256) void cif_scatter(
        const float* __restrict__ x, const float* __restrict__ csum,
        float* __restrict__ feats) {
    const int tid  = threadIdx.x;
    const int lane = tid & 31;
    const int wv   = tid >> 5;
    const int col  = blockIdx.x * 8 + wv;      // 0 .. B*S-1
    const int b    = col >> 12;
    const int s    = col & (S_DIM - 1);

    const float cc = csum[b * S_DIM + s];
    const float cp = (s > 0) ? csum[b * S_DIM + s - 1] : 0.f;
    const float alpha = cc - cp;
    const int dest1 = (int)floorf(cc + 0.0001f);
    const int dest0 = (s > 0) ? (int)floorf(cp + 0.0001f) : 0;
    const int fire  = dest1 - dest0;
    const float right_w = (fire > 0) ? (cc - (float)dest1) : alpha;
    const float extra   = (fire > 1) ? (float)(fire - 1) : 0.f;
    const float left_w  = alpha - right_w - extra;
    const int eidx      = (dest0 + 1 < T_MAX - 1) ? (dest0 + 1) : (T_MAX - 1);
    const int n_extra   = (fire > 2) ? (fire - 2) : 0;
    const float exw     = (extra < 1.0f) ? extra : 1.0f;

    const size_t xbase = ((size_t)s * B_DIM + b) * C_DIM;
    float* frow = feats + (size_t)b * T_MAX * C_DIM;

    for (int c = lane; c < C_DIM - 1; c += 32) {
        const float xv = x[xbase + c];
        if (dest1 < T_MAX && right_w != 0.f)
            atomicAdd(&frow[(size_t)dest1 * C_DIM + c], right_w * xv);
        if (dest0 < T_MAX && left_w != 0.f)
            atomicAdd(&frow[(size_t)dest0 * C_DIM + c], left_w * xv);
        if (extra > 0.f)
            atomicAdd(&frow[(size_t)eidx * C_DIM + c], exw * xv);
        for (int t = eidx + 1; t < eidx + 1 + n_extra && t < T_MAX; ++t)
            atomicAdd(&frow[(size_t)t * C_DIM + c], xv);  // BETA = 1.0
    }
}

// ---------------------------------------------------------------------------
// Kernel 5a: split W (512 x 511, row-major [co][k]) into bf16 hi/lo padded to
// K=512, layout [co][k].
// ---------------------------------------------------------------------------
__global__ __launch_bounds__(256) void cif_pack_w(
        const float* __restrict__ W,
        unsigned short* __restrict__ wh, unsigned short* __restrict__ wl) {
    const size_t i = ((size_t)blockIdx.x * 256 + threadIdx.x) * 4;
    #pragma unroll
    for (int j = 0; j < 4; ++j) {
        const size_t idx = i + j;
        const int n = (int)(idx >> 9);
        const int k = (int)(idx & 511);
        const float v = (k < C_DIM - 1) ? W[(size_t)n * (C_DIM - 1) + k] : 0.f;
        split_bf16(v, wh[idx], wl[idx]);
    }
}

// ---------------------------------------------------------------------------
// Kernel 5b: split feats (8192 x 512 fp32) into bf16 hi/lo, layout [m][k].
// ---------------------------------------------------------------------------
__global__ __launch_bounds__(256) void cif_pack_feats(
        const float* __restrict__ feats,
        unsigned short* __restrict__ fh, unsigned short* __restrict__ fl) {
    const size_t i = ((size_t)blockIdx.x * 256 + threadIdx.x) * 4;
    #pragma unroll
    for (int j = 0; j < 4; ++j)
        split_bf16(feats[i + j], fh[i + j], fl[i + j]);
}

// ---------------------------------------------------------------------------
// Kernel 6: out[t,b,co] = feats[b,t,:] @ W[co,:] + bias[co]
// M = 8192, N = 512, K = 512 (padded). Block = 128 threads (4 waves),
// tile 32(M) x 64(N); each wave owns two 16x16 accumulators.
// Operands pre-split to bf16 hi/lo; per K-step: async-stage tiles to LDS,
// 12 x ds_load_b128 fragment loads, 6 x v_wmma_f32_16x16x32_bf16
// (hi*hi + hi*lo + lo*hi per accumulator) for ~fp32 accuracy.
// ---------------------------------------------------------------------------
__global__ __launch_bounds__(128) void cif_gemm_wmma(
        const unsigned short* __restrict__ fh, const unsigned short* __restrict__ fl,
        const unsigned short* __restrict__ wh, const unsigned short* __restrict__ wl,
        const float* __restrict__ bias, float* __restrict__ out) {
    __shared__ __align__(16) unsigned short Ah[32 * 32];
    __shared__ __align__(16) unsigned short Al[32 * 32];
    __shared__ __align__(16) unsigned short Bh[64 * 32];
    __shared__ __align__(16) unsigned short Bl[64 * 32];

    const int tid  = threadIdx.x;
    const int lane = tid & 31;
    const int wv   = tid >> 5;                 // 0..3
    const int m0   = blockIdx.x * 32;
    const int n0g  = blockIdx.y * 64;

    const int half = (lane >> 4) << 3;         // 0 or 8
    const int rA0  = lane & 15;
    const int rA1  = 16 + (lane & 15);
    const int colB = (wv << 4) + (lane & 15);

    v8f acc0 = {}, acc1 = {};

    for (int kk = 0; kk < C_DIM; kk += 32) {
        // Stage A tiles (32x32 hi + lo): 256 x 16B chunks, 2 per thread.
        #pragma unroll
        for (int t = 0; t < 2; ++t) {
            const int c     = tid + t * 128;   // 0..255
            const int which = c >> 7;          // 0 = hi, 1 = lo
            const int idx   = c & 127;
            const int row   = idx >> 2;        // 0..31
            const int part  = idx & 3;         // 16B chunk in row
            const unsigned short* src = (which ? fl : fh)
                + (size_t)(m0 + row) * C_DIM + kk + part * 8;
            unsigned short* dst = (which ? Al : Ah) + row * 32 + part * 8;
            copy16_async(src, dst);
        }
        // Stage B tiles (64x32 hi + lo): 512 x 16B chunks, 4 per thread.
        #pragma unroll
        for (int t = 0; t < 4; ++t) {
            const int c     = tid + t * 128;   // 0..511
            const int which = c >> 8;
            const int idx   = c & 255;
            const int n     = idx >> 2;        // 0..63
            const int part  = idx & 3;
            const unsigned short* src = (which ? wl : wh)
                + (size_t)(n0g + n) * C_DIM + kk + part * 8;
            unsigned short* dst = (which ? Bl : Bh) + n * 32 + part * 8;
            copy16_async(src, dst);
        }
        async_copy_wait();
        __syncthreads();

        BF16x16 a0h, a0l, a1h, a1l, bhf, blf;
        load_frag(a0h, &Ah[rA0 * 32], half);
        load_frag(a0l, &Al[rA0 * 32], half);
        load_frag(a1h, &Ah[rA1 * 32], half);
        load_frag(a1l, &Al[rA1 * 32], half);
        load_frag(bhf, &Bh[colB * 32], half);
        load_frag(blf, &Bl[colB * 32], half);

        acc0 = __builtin_amdgcn_wmma_f32_16x16x32_bf16(false, a0h.v, false, bhf.v,
                                                       (short)0, acc0, false, false);
        acc0 = __builtin_amdgcn_wmma_f32_16x16x32_bf16(false, a0h.v, false, blf.v,
                                                       (short)0, acc0, false, false);
        acc0 = __builtin_amdgcn_wmma_f32_16x16x32_bf16(false, a0l.v, false, bhf.v,
                                                       (short)0, acc0, false, false);
        acc1 = __builtin_amdgcn_wmma_f32_16x16x32_bf16(false, a1h.v, false, bhf.v,
                                                       (short)0, acc1, false, false);
        acc1 = __builtin_amdgcn_wmma_f32_16x16x32_bf16(false, a1h.v, false, blf.v,
                                                       (short)0, acc1, false, false);
        acc1 = __builtin_amdgcn_wmma_f32_16x16x32_bf16(false, a1l.v, false, bhf.v,
                                                       (short)0, acc1, false, false);
        __syncthreads();
    }

    // C/D layout: VGPR r, lanes 0-15 -> M=r, N=lane; lanes 16-31 -> M=r+8.
    const int nl = lane & 15;
    const int co = n0g + (wv << 4) + nl;
    const float bv = bias[co];
    const int mhalf = (lane >> 4) << 3;
    #pragma unroll
    for (int r = 0; r < 8; ++r) {
        {
            const int m  = m0 + mhalf + r;
            const int bb = m >> 9;
            const int tt = m & (T_MAX - 1);
            out[((size_t)tt * B_DIM + bb) * C_DIM + co] = acc0[r] + bv;
        }
        {
            const int m  = m0 + 16 + mhalf + r;
            const int bb = m >> 9;
            const int tt = m & (T_MAX - 1);
            out[((size_t)tt * B_DIM + bb) * C_DIM + co] = acc1[r] + bv;
        }
    }
}

// ---------------------------------------------------------------------------
extern "C" void kernel_launch(void* const* d_in, const int* in_sizes, int n_in,
                              void* d_out, int out_size, void* d_ws, size_t ws_size,
                              hipStream_t stream) {
    (void)in_sizes; (void)n_in; (void)out_size; (void)ws_size;

    const float*         x    = (const float*)d_in[0];
    const unsigned char* mask = (const unsigned char*)d_in[1];
    const int*           tl   = (const int*)d_in[2];
    const float*         W    = (const float*)d_in[3];
    const float*         bias = (const float*)d_in[4];
    float* out = (float*)d_out;

    // Workspace layout (all 16B aligned).
    float* alpha_ws = (float*)d_ws;                                 // B*S
    float* csum_ws  = alpha_ws + (size_t)B_DIM * S_DIM;             // B*S
    float* feats_ws = csum_ws + (size_t)B_DIM * S_DIM;              // B*T*512
    float* asum_ws  = feats_ws + (size_t)B_DIM * T_MAX * C_DIM;     // B
    unsigned short* fh = (unsigned short*)(asum_ws + B_DIM);        // 8192*512
    unsigned short* fl = fh + (size_t)B_DIM * T_MAX * C_DIM;        // 8192*512
    unsigned short* whs = fl + (size_t)B_DIM * T_MAX * C_DIM;       // 512*512
    unsigned short* wls = whs + (size_t)C_DIM * C_DIM;              // 512*512
    float* out_asum = out + (size_t)T_MAX * B_DIM * C_DIM;

    cif_alpha<<<B_DIM, 256, 0, stream>>>(x, mask, alpha_ws, asum_ws, out_asum);
    cif_scan<<<B_DIM, 1024, 0, stream>>>(alpha_ws, asum_ws, tl, csum_ws);

    const int n4 = (B_DIM * T_MAX * C_DIM) / 4;
    cif_zero<<<(n4 + 255) / 256, 256, 0, stream>>>((float4*)feats_ws, n4);

    cif_pack_w<<<(C_DIM * C_DIM) / (256 * 4), 256, 0, stream>>>(W, whs, wls);

    cif_scatter<<<(B_DIM * S_DIM) / 8, 256, 0, stream>>>(x, csum_ws, feats_ws);

    cif_pack_feats<<<(B_DIM * T_MAX * C_DIM) / (256 * 4), 256, 0, stream>>>(
        feats_ws, fh, fl);

    dim3 gemm_grid((B_DIM * T_MAX) / 32, C_DIM / 64);               // 256 x 8
    cif_gemm_wmma<<<gemm_grid, 128, 0, stream>>>(fh, fl, whs, wls, bias, out);
}
